// QLinear_75058848465189
// MI455X (gfx1250) — 8-bit matrix operand order repaired and checked by reading, compiled, not run on hardware
//
#include <hip/hip_runtime.h>

typedef int   v8i __attribute__((ext_vector_type(8)));
typedef int   v4i __attribute__((ext_vector_type(4)));
typedef int   v2i __attribute__((ext_vector_type(2)));
typedef float v4f __attribute__((ext_vector_type(4)));

#define NROWS 32768
#define KDIM  768
#define NOUT  768

__global__ __launch_bounds__(256) void absmax_kernel(const float* __restrict__ w,
                                                     unsigned int* __restrict__ amax) {
    __shared__ float red[256];
    int g = blockIdx.x * 256 + threadIdx.x;
    float4 v = ((const float4*)w)[g];
    float m = fmaxf(fmaxf(fabsf(v.x), fabsf(v.y)), fmaxf(fabsf(v.z), fabsf(v.w)));
    red[threadIdx.x] = m;
    __syncthreads();
    for (int s = 128; s > 0; s >>= 1) {
        if (threadIdx.x < s) red[threadIdx.x] = fmaxf(red[threadIdx.x], red[threadIdx.x + s]);
        __syncthreads();
    }
    if (threadIdx.x == 0) atomicMax(amax, __float_as_uint(red[0]));
}

__device__ __forceinline__ int quant1(float v, float inv) {
    float r = rintf(v * inv);
    r = fminf(fmaxf(r, -128.0f), 127.0f);
    return (int)r;
}

__global__ __launch_bounds__(256) void wquant_kernel(const float* __restrict__ w,
                                                     const unsigned int* __restrict__ amax,
                                                     char* __restrict__ wq) {
    float am = __uint_as_float(*amax);
    float inv = 127.0f / am;
    int g = blockIdx.x * 256 + threadIdx.x;
    float4 v = ((const float4*)w)[g];
    int q0 = quant1(v.x, inv), q1 = quant1(v.y, inv);
    int q2 = quant1(v.z, inv), q3 = quant1(v.w, inv);
    ((int*)wq)[g] = (q0 & 255) | ((q1 & 255) << 8) | ((q2 & 255) << 16) | ((q3 & 255) << 24);
}

__global__ __launch_bounds__(256) void bias_kernel(const float* __restrict__ bias,
                                                   const float* __restrict__ a_s,
                                                   const unsigned int* __restrict__ amax,
                                                   int* __restrict__ bint,
                                                   float* __restrict__ bs_slot,
                                                   float* __restrict__ out_tail) {
    int i = blockIdx.x * 256 + threadIdx.x;
    float w_s = __uint_as_float(*amax) * (1.0f / 127.0f);
    float b_s = w_s * a_s[0];
    float r = rintf(bias[i] / b_s);
    r = fminf(fmaxf(r, -2.1474836e9f), 2.1474835e9f);
    bint[i] = (int)r;
    if (i == 0) { *bs_slot = b_s; *out_tail = b_s; }
}

__global__ __launch_bounds__(256) void qgemm_kernel(const float* __restrict__ x,
                                                    const float* __restrict__ a_s,
                                                    const char*  __restrict__ wq,
                                                    const int*   __restrict__ bint,
                                                    const float* __restrict__ bs_slot,
                                                    float* __restrict__ out) {
    __shared__ __align__(16) char lds_a[2][32 * 72];

    const int tid   = threadIdx.x;
    const int lane  = tid & 31;
    const int wave  = tid >> 5;
    const int l15   = lane & 15;
    const int lhalf = (lane >> 4) & 1;
    const int mblk  = blockIdx.x;

    const float inv_as = 1.0f / a_s[0];
    const float b_s    = *bs_slot;

    const int qrow = tid >> 3;
    const int qk   = (tid & 7) * 8;
    const float* xrow = x + (size_t)(mblk * 32 + qrow) * KDIM;

    v8i acc[2][6];
#pragma unroll
    for (int t = 0; t < 2; ++t)
#pragma unroll
        for (int j = 0; j < 6; ++j)
#pragma unroll
            for (int v = 0; v < 8; ++v) acc[t][j][v] = 0;

    int bfrag[6];
#pragma unroll
    for (int j = 0; j < 6; ++j) bfrag[j] = bint[wave * 96 + j * 16 + l15];

    {
        v4f va = __builtin_nontemporal_load((const v4f*)(xrow + 0 + qk));
        v4f vb = __builtin_nontemporal_load((const v4f*)(xrow + 0 + qk + 4));
        int lo = (quant1(va[0], inv_as) & 255)        | ((quant1(va[1], inv_as) & 255) << 8) |
                 ((quant1(va[2], inv_as) & 255) << 16) | ((quant1(va[3], inv_as) & 255) << 24);
        int hi = (quant1(vb[0], inv_as) & 255)        | ((quant1(vb[1], inv_as) & 255) << 8) |
                 ((quant1(vb[2], inv_as) & 255) << 16) | ((quant1(vb[3], inv_as) & 255) << 24);
        *(v2i*)(&lds_a[0][qrow * 72 + qk]) = (v2i){lo, hi};
    }
    __syncthreads();

    for (int ks = 0; ks < KDIM / 64; ++ks) {
        const int k0 = ks * 64;
        const int p  = ks & 1;

        v8i afrag[2];
#pragma unroll
        for (int t = 0; t < 2; ++t) {
            const char* ab = &lds_a[p][(t * 16 + l15) * 72 + lhalf * 16];
            v2i c0 = *(const v2i*)(ab +  0);
            v2i c1 = *(const v2i*)(ab +  8);
            v2i c2 = *(const v2i*)(ab + 32);
            v2i c3 = *(const v2i*)(ab + 40);
            afrag[t] = (v8i){c0[0], c0[1], c1[0], c1[1], c2[0], c2[1], c3[0], c3[1]};
        }

        if (ks + 1 < KDIM / 64) {
            v4f va = __builtin_nontemporal_load((const v4f*)(xrow + k0 + 64 + qk));
            v4f vb = __builtin_nontemporal_load((const v4f*)(xrow + k0 + 64 + qk + 4));
            int lo = (quant1(va[0], inv_as) & 255)        | ((quant1(va[1], inv_as) & 255) << 8) |
                     ((quant1(va[2], inv_as) & 255) << 16) | ((quant1(va[3], inv_as) & 255) << 24);
            int hi = (quant1(vb[0], inv_as) & 255)        | ((quant1(vb[1], inv_as) & 255) << 8) |
                     ((quant1(vb[2], inv_as) & 255) << 16) | ((quant1(vb[3], inv_as) & 255) << 24);
            *(v2i*)(&lds_a[1 - p][qrow * 72 + qk]) = (v2i){lo, hi};
        }

        v8i bf[6];
#pragma unroll
        for (int j = 0; j < 6; ++j) {
            const char* wb = wq + (size_t)(wave * 96 + j * 16 + l15) * KDIM + k0 + lhalf * 16;
            v4i blo = *(const v4i*)(wb);
            v4i bhi = *(const v4i*)(wb + 32);
            bf[j] = (v8i){blo[0], blo[1], blo[2], blo[3], bhi[0], bhi[1], bhi[2], bhi[3]};
        }

#pragma unroll
        for (int j = 0; j < 6; ++j) {
            acc[0][j] = __builtin_amdgcn_wmma_i32_16x16x64_iu8(true, afrag[0], true, bf[j],
                                                               acc[0][j], false, false);
            acc[1][j] = __builtin_amdgcn_wmma_i32_16x16x64_iu8(true, afrag[1], true, bf[j],
                                                               acc[1][j], false, false);
        }

        __syncthreads();
    }

#pragma unroll
    for (int t = 0; t < 2; ++t)
#pragma unroll
        for (int j = 0; j < 6; ++j) {
            int n = wave * 96 + j * 16 + l15;
#pragma unroll
            for (int v = 0; v < 8; ++v) {
                int m = mblk * 32 + t * 16 + lhalf * 8 + v;
                float val = (float)(acc[t][j][v] + bfrag[j]) * b_s;
                __builtin_nontemporal_store(val, &out[(size_t)m * NOUT + n]);
            }
        }
}

extern "C" void kernel_launch(void* const* d_in, const int* in_sizes, int n_in,
                              void* d_out, int out_size, void* d_ws, size_t ws_size,
                              hipStream_t stream) {
    const float* x    = (const float*)d_in[0];
    const float* a_s  = (const float*)d_in[1];
    const float* w    = (const float*)d_in[2];
    const float* bias = (const float*)d_in[3];
    float* out = (float*)d_out;

    unsigned int* amax = (unsigned int*)d_ws;
    float* bs_slot     = (float*)d_ws + 1;
    int*   bint        = (int*)((char*)d_ws + 64);
    char*  wq          = (char*)d_ws + 4096;

    hipMemsetAsync(d_ws, 0, 4, stream);
    absmax_kernel<<<576, 256, 0, stream>>>(w, amax);
    wquant_kernel<<<576, 256, 0, stream>>>(w, amax, wq);
    bias_kernel<<<3, 256, 0, stream>>>(bias, a_s, amax, bint, bs_slot,
                                       out + (size_t)NROWS * NOUT);
    qgemm_kernel<<<1024, 256, 0, stream>>>(x, a_s, wq, bint, bs_slot, out);
}
